// AttentionBlockB_43885975830637
// MI455X (gfx1250) — matrix operand order repaired and checked by reading, not compiled, not run on hardware
//
#include <hip/hip_runtime.h>

typedef __attribute__((ext_vector_type(16))) _Float16 v16h;
typedef __attribute__((ext_vector_type(8)))  float    v8f;

#define WMMA(a,b,c) __builtin_amdgcn_wmma_f32_16x16x32_f16(false,(a),false,(b),(short)0,(c),false,false)

// A fragment (16x32 f16): lane m = L%16, hi = L>>4.
// elements 0..15 = 16 contiguous K starting at k0 + hi*16
// caller passes p = rowbase(m) + k0 + hi*16   (16B aligned)
static __device__ __forceinline__ v16h load_afrag(const _Float16* p){
  v16h a;
  ((uint4*)&a)[0] = *(const uint4*)(p);
  ((uint4*)&a)[1] = *(const uint4*)(p + 8);
  return a;
}
// B fragment (32x16 f16), B stored transposed Bt[n][k]: lane n = L%16, hi = L>>4.
// elements 0..15 = 16 contiguous K starting at k0 + hi*16.
// caller passes p = Bt + n*ldb + k0 + hi*16
static __device__ __forceinline__ v16h load_bfrag(const _Float16* p){
  v16h b;
  ((uint4*)&b)[0] = *(const uint4*)(p);
  ((uint4*)&b)[1] = *(const uint4*)(p + 8);
  return b;
}

// ---------------- K0: f32 -> f16 conversion (weights keep [N][K] layout) ---
__global__ void k_cvt(const float* __restrict__ s, _Float16* __restrict__ d, int n){
  int i = blockIdx.x*256 + threadIdx.x;
  if (i < n) d[i] = (_Float16)s[i];
}

// ---------------- K1: BN1 + exact GELU, NCHW -> [pixel][C] f16 -------------
__global__ void k_bn1_gelu(const float* __restrict__ x,
                           const float* __restrict__ g, const float* __restrict__ bb,
                           const float* __restrict__ mm, const float* __restrict__ vv,
                           _Float16* __restrict__ y1){
  __shared__ _Float16 t[32][33];
  int tx = threadIdx.x & 31, ty = threadIdx.x >> 5;
  int hw0 = blockIdx.x * 32, c0 = blockIdx.y * 32, b = blockIdx.z;
  const float* xb = x + ((size_t)b*256 + c0)*3136 + hw0;
  #pragma unroll
  for (int i = 0; i < 4; i++){
    int c  = ty + i*8;
    int cg = c0 + c;
    float val = xb[(size_t)c*3136 + tx];
    float s  = g[cg] * rsqrtf(vv[cg] + 1e-5f);
    float yv = (val - mm[cg]) * s + bb[cg];
    float ge = 0.5f * yv * (1.0f + erff(yv * 0.70710678118f));
    t[c][tx] = (_Float16)ge;
  }
  __syncthreads();
  _Float16* yp = y1 + ((size_t)b*3136 + hw0)*256 + c0;
  #pragma unroll
  for (int i = 0; i < 4; i++){
    int r = ty + i*8;                 // local pixel row
    yp[(size_t)r*256 + tx] = t[tx][r];
  }
}

// ---------------- K2: y2 = y1(100352x256) x conv_w^T + BN2 -> windowed -----
// 8 waves/block; each wave: 1 M-tile x 4 N-tiles, double-buffered k-loop.
__global__ __launch_bounds__(256)
void k_gemm_bn2(const _Float16* __restrict__ A, const _Float16* __restrict__ Bt,
                const float* __restrict__ g, const float* __restrict__ bb,
                const float* __restrict__ mm, const float* __restrict__ vv,
                _Float16* __restrict__ yw){
  int lane = threadIdx.x & 31, wv = threadIdx.x >> 5;
  int ln = lane & 15, hi = lane >> 4;
  int mbase = blockIdx.x*128 + wv*16;
  int nbase = blockIdx.y*64;
  v8f acc[4] = {};
  const _Float16* pa = A + (size_t)(mbase + ln)*256 + hi*16;
  const _Float16* pb[4];
  #pragma unroll
  for (int t = 0; t < 4; t++)
    pb[t] = Bt + (size_t)(nbase + t*16 + ln)*256 + hi*16;

  v16h a = load_afrag(pa);
  v16h b[4];
  #pragma unroll
  for (int t = 0; t < 4; t++) b[t] = load_bfrag(pb[t]);

  #pragma unroll
  for (int k0 = 0; k0 < 256; k0 += 32){
    v16h ca = a;
    v16h cb[4] = {b[0], b[1], b[2], b[3]};
    if (k0 + 32 < 256){               // prefetch next k-step before the WMMAs
      a = load_afrag(pa + k0 + 32);
      #pragma unroll
      for (int t = 0; t < 4; t++) b[t] = load_bfrag(pb[t] + k0 + 32);
    }
    #pragma unroll
    for (int t = 0; t < 4; t++)
      acc[t] = WMMA(ca, cb[t], acc[t]);
  }

  #pragma unroll
  for (int t = 0; t < 4; t++){
    int n = nbase + t*16 + ln;
    float s    = g[n] * rsqrtf(vv[n] + 1e-5f);
    float bias = bb[n] - mm[n]*s;
    #pragma unroll
    for (int r = 0; r < 8; r++){
      int m = mbase + r + 8*hi;               // global pixel
      float val = acc[t][r]*s + bias;
      int bidx = m / 3136, rem = m % 3136;
      int h = rem / 56, w = rem % 56;
      int wi  = bidx*64 + (h/7)*8 + (w/7);
      int pos = (h%7)*7 + (w%7);
      yw[((size_t)wi*49 + pos)*256 + n] = (_Float16)val;
    }
  }
}

// ---------------- K3: per-(window,head) attention, 1 wave / workgroup ------
__global__ __launch_bounds__(32)
void k_attn(const _Float16* __restrict__ yw,
            const _Float16* __restrict__ wq,
            const _Float16* __restrict__ wkv,
            const float* __restrict__ pos_emb,
            _Float16* __restrict__ aout){
  __shared__ _Float16 Qb[64*64];   // q, later attn (row-major [i][d] / [i][j])
  __shared__ _Float16 Kb[64*64];   // k row-major [j][d]  == Bt for q.k^T
  __shared__ _Float16 Vt[64*64];   // v transposed [d][j] == Bt for attn.v
  __shared__ float pe[169];
  int lane = threadIdx.x;
  int ln = lane & 15, hi = lane >> 4;
  int wi = blockIdx.x;      // window 0..2047
  int head = blockIdx.y;    // head 0..7
  for (int t = lane; t < 169; t += 32) pe[t] = pos_emb[t];
  __syncthreads();

  const _Float16* Abase = yw + (size_t)wi*49*256;
  const _Float16* Wq = wq  + (size_t)head*64*256;
  const _Float16* Wk = wkv + (size_t)head*64*256;
  const _Float16* Wv = wkv + (size_t)(512 + head*64)*256;

  // ---- QKV projections (rows 49..63 zero-padded) ----
  for (int i0 = 0; i0 < 64; i0 += 16){
    int row = i0 + ln;
    bool valid = row < 49;
    v8f aq[4] = {}; v8f ak[4] = {}; v8f av[4] = {};
    const _Float16* pa = Abase + (size_t)row*256 + hi*16;
    v16h a = {};
    if (valid) a = load_afrag(pa);
    #pragma unroll
    for (int k0 = 0; k0 < 256; k0 += 32){
      // B fragments loaded in pairs to bound register pressure
      #pragma unroll
      for (int dp = 0; dp < 2; dp++){
        v16h b0 = load_bfrag(Wq + (size_t)((dp*2+0)*16 + ln)*256 + k0 + hi*16);
        v16h b1 = load_bfrag(Wq + (size_t)((dp*2+1)*16 + ln)*256 + k0 + hi*16);
        aq[dp*2+0] = WMMA(a, b0, aq[dp*2+0]);
        aq[dp*2+1] = WMMA(a, b1, aq[dp*2+1]);
      }
      #pragma unroll
      for (int dp = 0; dp < 2; dp++){
        v16h b0 = load_bfrag(Wk + (size_t)((dp*2+0)*16 + ln)*256 + k0 + hi*16);
        v16h b1 = load_bfrag(Wk + (size_t)((dp*2+1)*16 + ln)*256 + k0 + hi*16);
        ak[dp*2+0] = WMMA(a, b0, ak[dp*2+0]);
        ak[dp*2+1] = WMMA(a, b1, ak[dp*2+1]);
      }
      v16h anext = {};
      if (valid && k0 + 32 < 256) anext = load_afrag(pa + k0 + 32);
      #pragma unroll
      for (int dp = 0; dp < 2; dp++){
        v16h b0 = load_bfrag(Wv + (size_t)((dp*2+0)*16 + ln)*256 + k0 + hi*16);
        v16h b1 = load_bfrag(Wv + (size_t)((dp*2+1)*16 + ln)*256 + k0 + hi*16);
        av[dp*2+0] = WMMA(a, b0, av[dp*2+0]);
        av[dp*2+1] = WMMA(a, b1, av[dp*2+1]);
      }
      a = anext;
    }
    #pragma unroll
    for (int dt = 0; dt < 4; dt++){
      int d = dt*16 + ln;
      #pragma unroll
      for (int r = 0; r < 8; r++){
        int m = i0 + r + 8*hi;
        Qb[m*64 + d] = (_Float16)aq[dt][r];
        Kb[m*64 + d] = (_Float16)ak[dt][r];
        Vt[d*64 + m] = (_Float16)av[dt][r];
      }
    }
  }
  __syncthreads();

  // ---- dots = q.k^T * scale + mask; softmax; attn overwrites Qb ----
  for (int i0 = 0; i0 < 64; i0 += 16){
    v8f dts[4] = {};
    #pragma unroll
    for (int k0 = 0; k0 < 64; k0 += 32){
      v16h a = load_afrag(Qb + (i0 + ln)*64 + k0 + hi*16);
      v16h bfr[4];
      #pragma unroll
      for (int jt = 0; jt < 4; jt++)
        bfr[jt] = load_bfrag(Kb + (jt*16 + ln)*64 + k0 + hi*16);
      #pragma unroll
      for (int jt = 0; jt < 4; jt++) dts[jt] = WMMA(a, bfr[jt], dts[jt]);
    }
    #pragma unroll
    for (int r = 0; r < 8; r++){
      int i  = i0 + r + 8*hi;
      int ic = i < 49 ? i : 48;
      int a1 = ic / 7, b1 = ic % 7;
      float vrow[4];
      #pragma unroll
      for (int jt = 0; jt < 4; jt++){
        int j = jt*16 + ln;
        float val = -3.0e38f;
        if (j < 49){
          int a2 = j / 7, b2 = j % 7;
          val = dts[jt][r]*0.125f + pe[(a2 - a1 + 6)*13 + (b2 - b1 + 6)];
        }
        vrow[jt] = val;
      }
      float mx = fmaxf(fmaxf(vrow[0], vrow[1]), fmaxf(vrow[2], vrow[3]));
      #pragma unroll
      for (int s = 1; s < 16; s <<= 1) mx = fmaxf(mx, __shfl_xor(mx, s, 32));
      float sum = 0.f;
      #pragma unroll
      for (int jt = 0; jt < 4; jt++){ vrow[jt] = __expf(vrow[jt] - mx); sum += vrow[jt]; }
      #pragma unroll
      for (int s = 1; s < 16; s <<= 1) sum += __shfl_xor(sum, s, 32);
      float inv = 1.0f / sum;
      #pragma unroll
      for (int jt = 0; jt < 4; jt++)
        Qb[i*64 + jt*16 + ln] = (_Float16)(vrow[jt]*inv);
    }
  }
  __syncthreads();

  // ---- out = attn x v ----
  for (int i0 = 0; i0 < 64; i0 += 16){
    v8f oo[4] = {};
    #pragma unroll
    for (int k0 = 0; k0 < 64; k0 += 32){
      v16h a = load_afrag(Qb + (i0 + ln)*64 + k0 + hi*16);
      v16h bfr[4];
      #pragma unroll
      for (int dt = 0; dt < 4; dt++)
        bfr[dt] = load_bfrag(Vt + (dt*16 + ln)*64 + k0 + hi*16);
      #pragma unroll
      for (int dt = 0; dt < 4; dt++) oo[dt] = WMMA(a, bfr[dt], oo[dt]);
    }
    #pragma unroll
    for (int dt = 0; dt < 4; dt++){
      int d = head*64 + dt*16 + ln;
      #pragma unroll
      for (int r = 0; r < 8; r++){
        int i = i0 + r + 8*hi;
        if (i < 49) aout[((size_t)wi*49 + i)*512 + d] = (_Float16)oo[dt][r];
      }
    }
  }
}

// ---------------- K4: out-proj GEMM + bias + residual, scatter to NCHW -----
__global__ __launch_bounds__(256)
void k_gemm_out(const _Float16* __restrict__ A, const _Float16* __restrict__ Bt,
                const float* __restrict__ ob, const float* __restrict__ x,
                float* __restrict__ out){
  int lane = threadIdx.x & 31, wv = threadIdx.x >> 5;
  int ln = lane & 15, hi = lane >> 4;
  int mbase = blockIdx.x*128 + wv*16;
  int nbase = blockIdx.y*64;
  v8f acc[4] = {};
  const _Float16* pa = A + (size_t)(mbase + ln)*512 + hi*16;
  const _Float16* pb[4];
  #pragma unroll
  for (int t = 0; t < 4; t++)
    pb[t] = Bt + (size_t)(nbase + t*16 + ln)*512 + hi*16;

  v16h a = load_afrag(pa);
  v16h b[4];
  #pragma unroll
  for (int t = 0; t < 4; t++) b[t] = load_bfrag(pb[t]);

  #pragma unroll
  for (int k0 = 0; k0 < 512; k0 += 32){
    v16h ca = a;
    v16h cb[4] = {b[0], b[1], b[2], b[3]};
    if (k0 + 32 < 512){
      a = load_afrag(pa + k0 + 32);
      #pragma unroll
      for (int t = 0; t < 4; t++) b[t] = load_bfrag(pb[t] + k0 + 32);
    }
    #pragma unroll
    for (int t = 0; t < 4; t++)
      acc[t] = WMMA(ca, cb[t], acc[t]);
  }

  #pragma unroll
  for (int t = 0; t < 4; t++){
    int n = nbase + t*16 + ln;
    float bias = ob[n];
    #pragma unroll
    for (int r = 0; r < 8; r++){
      int m = mbase + r + 8*hi;           // window-pixel row
      int wi = m / 49, pos = m % 49;
      int b_ = wi / 64, wrem = wi % 64;
      int h = (wrem/8)*7 + pos/7;
      int w = (wrem%8)*7 + pos%7;
      size_t oidx = ((size_t)b_*256 + n)*3136 + (size_t)h*56 + w;
      out[oidx] = acc[t][r] + bias + x[oidx];
    }
  }
}

extern "C" void kernel_launch(void* const* d_in, const int* in_sizes, int n_in,
                              void* d_out, int out_size, void* d_ws, size_t ws_size,
                              hipStream_t stream){
  (void)in_sizes; (void)n_in; (void)out_size; (void)ws_size;
  const float* x      = (const float*)d_in[0];
  const float* bn1_g  = (const float*)d_in[1];
  const float* bn1_b  = (const float*)d_in[2];
  const float* bn1_m  = (const float*)d_in[3];
  const float* bn1_v  = (const float*)d_in[4];
  const float* conv_w = (const float*)d_in[5];
  const float* bn2_g  = (const float*)d_in[6];
  const float* bn2_b  = (const float*)d_in[7];
  const float* bn2_m  = (const float*)d_in[8];
  const float* bn2_v  = (const float*)d_in[9];
  const float* pos_emb= (const float*)d_in[10];
  const float* q_w    = (const float*)d_in[11];
  const float* kv_w   = (const float*)d_in[12];
  const float* out_w  = (const float*)d_in[13];
  const float* out_b  = (const float*)d_in[14];
  float* out = (float*)d_out;

  char* ws = (char*)d_ws;
  size_t off = 0;
  auto alloc = [&](size_t halfs)->_Float16* {
    _Float16* p = (_Float16*)(ws + off);
    off += halfs * sizeof(_Float16);
    off = (off + 255) & ~(size_t)255;
    return p;
  };
  _Float16* wc  = alloc((size_t)256*256);     // conv weight f16  [N][K]
  _Float16* wq  = alloc((size_t)512*256);     // q weight f16
  _Float16* wkv = alloc((size_t)1024*256);    // kv weight f16
  _Float16* wo  = alloc((size_t)256*512);     // out weight f16
  _Float16* y1  = alloc((size_t)100352*256);  // gelu(bn1(x)) [pixel][C]
  _Float16* yw  = alloc((size_t)100352*256);  // bn2(conv(y1)) windowed [win][49][C]
  _Float16* ao  = alloc((size_t)100352*512);  // attention out [win][49][inner]

  k_cvt<<<(256*256 + 255)/256, 256, 0, stream>>>(conv_w, wc, 256*256);
  k_cvt<<<(512*256 + 255)/256, 256, 0, stream>>>(q_w,    wq, 512*256);
  k_cvt<<<(1024*256 + 255)/256,256, 0, stream>>>(kv_w,  wkv, 1024*256);
  k_cvt<<<(256*512 + 255)/256, 256, 0, stream>>>(out_w,  wo, 256*512);

  k_bn1_gelu<<<dim3(98, 8, 32), 256, 0, stream>>>(x, bn1_g, bn1_b, bn1_m, bn1_v, y1);
  k_gemm_bn2<<<dim3(784, 4), 256, 0, stream>>>(y1, wc, bn2_g, bn2_b, bn2_m, bn2_v, yw);
  k_attn<<<dim3(2048, 8), 32, 0, stream>>>(yw, wq, wkv, pos_emb, ao);
  k_gemm_out<<<dim3(784, 4), 256, 0, stream>>>(ao, wo, out_b, x, out);
}
